// MultiHeadPositionBiasBasedForMLM_1125281432113
// MI455X (gfx1250) — matrix operand order repaired and checked by reading, not compiled, not run on hardware
//
#include <hip/hip_runtime.h>
#include <math.h>

typedef __attribute__((ext_vector_type(16))) __bf16 v16bf;
typedef __attribute__((ext_vector_type(8)))  float  v8f;
typedef __attribute__((ext_vector_type(4)))  int    v4i;
typedef unsigned short u16;
typedef __attribute__((ext_vector_type(8))) unsigned short u16x8;

#define B_  8
#define H_  12
#define Q_  512
#define S_  512
#define D_  768
#define HD_ 64
#define P_  200
#define V_  30522

#define AS1 __attribute__((address_space(1)))
#define AS3 __attribute__((address_space(3)))

typedef AS1 v4i* gptr128;   // global 16B chunk pointer
typedef AS3 v4i* lptr128;   // LDS 16B chunk pointer

#if __has_builtin(__builtin_amdgcn_global_load_async_to_lds_b128)
#define HAVE_ASYNC 1
#else
#define HAVE_ASYNC 0
#endif

// ---------- helpers ----------
static __device__ __forceinline__ u16 f2bf(float f) {
  unsigned u = __builtin_bit_cast(unsigned, f);
  unsigned r = u + 0x7FFFu + ((u >> 16) & 1u);   // round-to-nearest-even
  return (u16)(r >> 16);
}
static __device__ __forceinline__ v16bf mkfrag(u16x8 lo, u16x8 hi) {
  union { u16x8 h[2]; v16bf v; } u;
  u.h[0] = lo; u.h[1] = hi;
  return u.v;
}
// A-matrix fragment: lane holds row; elems 0..15 = K[off..off+15]
static __device__ __forceinline__ v16bf loadA16(const u16* row, int off) {
  return mkfrag(*(const u16x8*)(row + off), *(const u16x8*)(row + off + 8));
}
// B-matrix fragment: lane holds column; elems 0..15 = K[kb..kb+15] (K-major contiguous)
static __device__ __forceinline__ v16bf loadB16(const u16* p) {
  return mkfrag(*(const u16x8*)p, *(const u16x8*)(p + 8));
}
static __device__ __forceinline__ v8f wmma_bf16(v16bf a, v16bf b, v8f c) {
  return __builtin_amdgcn_wmma_f32_16x16x32_bf16(false, a, false, b, (short)0, c, false, false);
}
static __device__ __forceinline__ v8f vzero() {
  v8f z = {0.f,0.f,0.f,0.f,0.f,0.f,0.f,0.f};
  return z;
}
static __device__ __forceinline__ void wait_async0() {
#if __has_builtin(__builtin_amdgcn_s_wait_asynccnt)
  __builtin_amdgcn_s_wait_asynccnt(0);
#else
  asm volatile("s_wait_asynccnt 0" ::: "memory");
#endif
}
// Stage `n8` 16-byte chunks from global `src` to LDS `dst` cooperatively (blockDim.x threads).
static __device__ __forceinline__ void stage_tile(const u16* __restrict__ src,
                                                  u16* __restrict__ dst, int n8) {
#if HAVE_ASYNC
  for (int c = threadIdx.x; c < n8; c += blockDim.x)
    __builtin_amdgcn_global_load_async_to_lds_b128(
        (gptr128)(u16*)(src + c * 8), (lptr128)(dst + c * 8), 0, 0);
  wait_async0();
#else
  for (int c = threadIdx.x; c < n8; c += blockDim.x)
    *(u16x8*)(dst + c * 8) = *(const u16x8*)(src + c * 8);
#endif
}

// ---------- conversion kernels ----------
__global__ void cvt_f32_bf16(const float* __restrict__ in, u16* __restrict__ out, long n) {
  for (long i = (long)blockIdx.x * blockDim.x + threadIdx.x; i < n;
       i += (long)gridDim.x * blockDim.x)
    out[i] = f2bf(in[i]);
}

// v[b,s,h*64+dc] -> vt[((b*H+h)*64+dc)*S + s]   (K-major for alpha@V)
__global__ void cvt_v_transpose(const float* __restrict__ in, u16* __restrict__ out) {
  const long N = (long)B_ * S_ * D_;
  for (long i = (long)blockIdx.x * blockDim.x + threadIdx.x; i < N;
       i += (long)gridDim.x * blockDim.x) {
    long d = i % D_;
    long s = (i / D_) % S_;
    long b = i / ((long)D_ * S_);
    long h = d / HD_, dc = d % HD_;
    out[((b * H_ + h) * HD_ + dc) * S_ + s] = f2bf(in[i]);
  }
}

// W[d,n] -> WT[n*768 + d]   (K-major for dense GEMM B operand)
__global__ void cvt_w_transpose(const float* __restrict__ in, u16* __restrict__ out) {
  const long N = (long)D_ * D_;
  for (long i = (long)blockIdx.x * blockDim.x + threadIdx.x; i < N;
       i += (long)gridDim.x * blockDim.x) {
    long d = i / D_, n = i % D_;
    out[n * D_ + d] = f2bf(in[i]);
  }
}

// pos_q[q,d] = pos_emb[q,:200] @ pos_proj_w[:,d] + b[d]   (tiny GEMM, scalar FMA)
__global__ void posq_kernel(const float* __restrict__ pe, const float* __restrict__ w,
                            const float* __restrict__ bias, u16* __restrict__ out) {
  int q = blockIdx.x;
  for (int d = threadIdx.x; d < D_; d += blockDim.x) {
    float acc = bias[d];
    for (int p = 0; p < P_; ++p)
      acc = fmaf(pe[q * P_ + p], w[p * D_ + d], acc);
    out[q * D_ + d] = f2bf(acc);
  }
}

// ---------- fused attention: scores(WMMA) -> softmax -> alpha@V(WMMA) ----------
__global__ __launch_bounds__(32)
void attn_kernel(const float* __restrict__ src_scores, const float* __restrict__ seq_mask,
                 const u16* __restrict__ kq, const u16* __restrict__ vt,
                 const u16* __restrict__ posq, u16* __restrict__ seqsum) {
  __shared__ alignas(16) float sc[16][512];   // fp32 scores / exp values
  __shared__ alignas(16) u16   al[16][512];   // bf16 alpha (A-operand staging)

  int bid = blockIdx.x;           // B*H*(Q/16) = 3072
  int qt  = bid & 31;
  int h   = (bid >> 5) % H_;
  int b   = bid / (32 * H_);
  int q0  = qt * 16;

  int L      = threadIdx.x;
  int hrow   = L >> 4;            // 0 or 1 (lane half)
  int lane16 = L & 15;

  // pos_q A fragments for this q-tile / head (16 x 64, two K=32 steps)
  const u16* arow = posq + (size_t)(q0 + lane16) * D_ + h * HD_;
  v16bf a0 = loadA16(arow, 0  + hrow * 16);
  v16bf a1 = loadA16(arow, 32 + hrow * 16);

  const float* srcb = src_scores + (((size_t)b * H_ + h) * Q_) * S_;

  // ---- scores = pos_q @ k^T + src + mask ----
  for (int s0 = 0; s0 < S_; s0 += 16) {
    int s = s0 + lane16;
    const u16* krow = kq + ((size_t)b * S_ + s) * D_ + h * HD_;   // B column: d contiguous
    v16bf b0 = loadB16(krow + 0  + hrow * 16);
    v16bf b1 = loadB16(krow + 32 + hrow * 16);
    v8f acc = vzero();
    acc = wmma_bf16(a0, b0, acc);
    acc = wmma_bf16(a1, b1, acc);
    float mk = seq_mask[b * S_ + s] * (-10000.0f);
#pragma unroll
    for (int i = 0; i < 8; ++i) {
      int m = i + hrow * 8;
      sc[m][s] = acc[i] + srcb[(size_t)(q0 + m) * S_ + s] + mk;
    }
  }
  __syncthreads();

  // ---- softmax over S (2 lanes per row) ----
  {
    int r = L >> 1, half = L & 1;
    int j0 = half * 256, j1 = j0 + 256;
    float mx = -3.4e38f;
    for (int j = j0; j < j1; ++j) mx = fmaxf(mx, sc[r][j]);
    mx = fmaxf(mx, __shfl_xor(mx, 1, 32));
    float sum = 0.f;
    for (int j = j0; j < j1; ++j) { float e = expf(sc[r][j] - mx); sc[r][j] = e; sum += e; }
    sum += __shfl_xor(sum, 1, 32);
    float inv = 1.0f / sum;
    for (int j = j0; j < j1; ++j) al[r][j] = f2bf(sc[r][j] * inv);
  }
  __syncthreads();

  // ---- seq_sum = alpha @ v  (16 x 64, K = 512) ----
  v8f oacc[4];
#pragma unroll
  for (int t = 0; t < 4; ++t) oacc[t] = vzero();

  const u16* vbase = vt + (((size_t)b * H_ + h) * HD_) * S_;   // [d][s], s contiguous
  for (int ks = 0; ks < S_; ks += 32) {
    v16bf af = loadA16(&al[lane16][0], ks + hrow * 16);
    int kb = ks + hrow * 16;
#pragma unroll
    for (int t = 0; t < 4; ++t) {
      int d = t * 16 + lane16;
      v16bf bf_ = loadB16(vbase + (size_t)d * S_ + kb);
      oacc[t] = wmma_bf16(af, bf_, oacc[t]);
    }
  }
#pragma unroll
  for (int t = 0; t < 4; ++t)
#pragma unroll
    for (int i = 0; i < 8; ++i) {
      int m = i + hrow * 8;
      seqsum[((size_t)b * Q_ + q0 + m) * D_ + h * HD_ + t * 16 + lane16] = f2bf(oacc[t][i]);
    }
}

// ---------- dense (WMMA) + GELU(erf) + LayerNorm ----------
__global__ __launch_bounds__(256)
void dense_gelu_ln(const u16* __restrict__ seqsum, const u16* __restrict__ dwT,
                   const float* __restrict__ dbias, const float* __restrict__ lng,
                   const float* __restrict__ lnb, u16* __restrict__ hdn) {
  __shared__ alignas(16) float hx[16][768];   // phase 1: bf16 A-tile stage; phase 2: fp32 gelu
  __shared__ float psum[16][16], psq[16][16];
  __shared__ float mu[16], rstd[16];

  int m0 = blockIdx.x * 16;
  int L = threadIdx.x & 31, w = threadIdx.x >> 5;
  int hrow = L >> 4, lane16 = L & 15;

  // ---- phase 1: async-stage A tile (16x768 bf16 = 24 KB) into LDS ----
  u16* atile = (u16*)&hx[0][0];
  stage_tile(seqsum + (size_t)m0 * D_, atile, 16 * D_ / 8);
  __syncthreads();

  v8f acc[6];
#pragma unroll
  for (int t = 0; t < 6; ++t) acc[t] = vzero();

  const u16* arow = atile + (size_t)lane16 * D_;
  for (int ks = 0; ks < D_; ks += 32) {
    v16bf af = loadA16(arow, ks + hrow * 16);
    int kb = ks + hrow * 16;
#pragma unroll
    for (int t = 0; t < 6; ++t) {
      int n = w * 96 + t * 16 + lane16;
      v16bf bf_ = loadB16(dwT + (size_t)n * D_ + kb);   // W^T: d contiguous
      acc[t] = wmma_bf16(af, bf_, acc[t]);
    }
  }
  __syncthreads();   // all waves done reading A tile before hx is overwritten

  // ---- phase 2: bias + exact GELU into hx ----
#pragma unroll
  for (int t = 0; t < 6; ++t) {
    int n = w * 96 + t * 16 + lane16;
    float bb = dbias[n];
#pragma unroll
    for (int i = 0; i < 8; ++i) {
      int m = i + hrow * 8;
      float x = acc[t][i] + bb;
      hx[m][n] = 0.5f * x * (1.0f + erff(x * 0.70710678118654752f));
    }
  }
  __syncthreads();

  // ---- phase 3: LayerNorm ----
  int rr = threadIdx.x >> 4, cc = threadIdx.x & 15;   // 16 threads per row, 48 elems each
  float s1 = 0.f, s2 = 0.f;
  for (int j = cc * 48; j < cc * 48 + 48; ++j) { float x = hx[rr][j]; s1 += x; s2 += x * x; }
  psum[rr][cc] = s1; psq[rr][cc] = s2;
  __syncthreads();
  if (threadIdx.x < 16) {
    float a = 0.f, b = 0.f;
    for (int c = 0; c < 16; ++c) { a += psum[threadIdx.x][c]; b += psq[threadIdx.x][c]; }
    float m = a / (float)D_;
    float var = b / (float)D_ - m * m;
    mu[threadIdx.x] = m;
    rstd[threadIdx.x] = rsqrtf(var + 1e-12f);
  }
  __syncthreads();
  for (int j = cc * 48; j < cc * 48 + 48; ++j) {
    float y = (hx[rr][j] - mu[rr]) * rstd[rr] * lng[j] + lnb[j];
    hdn[(size_t)(m0 + rr) * D_ + j] = f2bf(y);
  }
}

// ---------- decoder GEMM: out[4096, 30522] = hdn @ dec_w^T + dec_b ----------
__global__ __launch_bounds__(256)
void decoder_gemm(const u16* __restrict__ hdn, const u16* __restrict__ decw,
                  const float* __restrict__ decb, float* __restrict__ out) {
  __shared__ alignas(16) u16 atile[32 * 768];   // 48 KB A tile, shared by all 8 waves

  int L = threadIdx.x & 31, w = threadIdx.x >> 5;
  int hrow = L >> 4, lane16 = L & 15;
  int m0   = blockIdx.y * 32;
  int mloc = (w & 1) * 16;                      // wave's 16-row group within the tile
  int nq   = w >> 1;                            // 4 col-groups of 64
  int nbase = blockIdx.x * 256 + nq * 64;

  // ---- async-stage A tile (32x768 bf16 = 48 KB) into LDS ----
  stage_tile(hdn + (size_t)m0 * D_, atile, 32 * D_ / 8);
  __syncthreads();

  v8f acc[4];
#pragma unroll
  for (int t = 0; t < 4; ++t) acc[t] = vzero();

  const u16* arow = atile + (size_t)(mloc + lane16) * D_;
  const u16* brow[4];
  int vraw[4];
#pragma unroll
  for (int t = 0; t < 4; ++t) {
    vraw[t] = nbase + t * 16 + lane16;
    int vcl = vraw[t] < V_ ? vraw[t] : (V_ - 1);
    brow[t] = decw + (size_t)vcl * D_;          // dec_w[v, :] -> K-major contiguous
  }

  for (int ks = 0; ks < D_; ks += 32) {
    int kb = ks + hrow * 16;
    __builtin_prefetch(brow[0] + kb + 128, 0, 1);   // global_prefetch on weight stream
    v16bf af = loadA16(arow, ks + hrow * 16);
#pragma unroll
    for (int t = 0; t < 4; ++t) {
      v16bf bf_ = loadB16(brow[t] + kb);
      acc[t] = wmma_bf16(af, bf_, acc[t]);
    }
  }
#pragma unroll
  for (int t = 0; t < 4; ++t) {
    if (vraw[t] < V_) {
      float bb = decb[vraw[t]];
#pragma unroll
      for (int i = 0; i < 8; ++i) {
        int m = m0 + mloc + i + hrow * 8;
        out[(size_t)m * V_ + vraw[t]] = acc[t][i] + bb;
      }
    }
  }
}

// ---------- launch ----------
extern "C" void kernel_launch(void* const* d_in, const int* in_sizes, int n_in,
                              void* d_out, int out_size, void* d_ws, size_t ws_size,
                              hipStream_t stream) {
  const float* src_scores = (const float*)d_in[0];
  const float* seq_k      = (const float*)d_in[1];
  const float* seq_v      = (const float*)d_in[2];
  const float* seq_mask   = (const float*)d_in[3];
  const float* pos_emb    = (const float*)d_in[4];
  const float* pos_w      = (const float*)d_in[5];
  const float* pos_b      = (const float*)d_in[6];
  const float* dense_w    = (const float*)d_in[7];
  const float* dense_b    = (const float*)d_in[8];
  const float* ln_g       = (const float*)d_in[9];
  const float* ln_b       = (const float*)d_in[10];
  const float* dec_w      = (const float*)d_in[11];
  const float* dec_b      = (const float*)d_in[12];
  float* out = (float*)d_out;

  // workspace layout (bf16 buffers, 256B-aligned offsets; ~74 MB total)
  char* ws = (char*)d_ws;
  u16* kq     = (u16*)(ws + 0);          //  6,291,456 B : K [B,S,D]
  u16* vt     = (u16*)(ws + 6291456);    //  6,291,456 B : V^T [B,H,64,S]
  u16* posq   = (u16*)(ws + 12582912);   //    786,432 B : pos_q [Q,D]
  u16* dwT    = (u16*)(ws + 13369344);   //  1,179,648 B : dense_w^T [D,D]
  u16* decwq  = (u16*)(ws + 14548992);   // 46,881,792 B : dec_w [V,D]
  u16* seqsum = (u16*)(ws + 61430784);   //  6,291,456 B : attention out [B,Q,D]
  u16* hdn    = (u16*)(ws + 67722240);   //  6,291,456 B : transform out [B,Q,D]

  cvt_f32_bf16   <<<dim3(4096), dim3(256), 0, stream>>>(seq_k, kq, (long)B_ * S_ * D_);
  cvt_v_transpose<<<dim3(4096), dim3(256), 0, stream>>>(seq_v, vt);
  cvt_w_transpose<<<dim3(2304), dim3(256), 0, stream>>>(dense_w, dwT);
  cvt_f32_bf16   <<<dim3(8192), dim3(256), 0, stream>>>(dec_w, decwq, (long)V_ * D_);
  posq_kernel    <<<dim3(Q_),   dim3(256), 0, stream>>>(pos_emb, pos_w, pos_b, posq);

  attn_kernel    <<<dim3(B_ * H_ * (Q_ / 16)), dim3(32), 0, stream>>>(
      src_scores, seq_mask, kq, vt, posq, seqsum);

  dense_gelu_ln  <<<dim3((B_ * Q_) / 16), dim3(256), 0, stream>>>(
      seqsum, dwT, dense_b, ln_g, ln_b, hdn);

  decoder_gemm   <<<dim3((V_ + 255) / 256, (B_ * Q_) / 32), dim3(256), 0, stream>>>(
      hdn, decwq, dec_b, out);
}
